// LlamaAttention_15977278341829
// MI455X (gfx1250) — compile-verified, matrix operand order checked by reading, not run on hardware
//
#include <hip/hip_runtime.h>
#include <hip/hip_bf16.h>

// ---------------------------------------------------------------------------
// CDNA5 / gfx1250 Llama attention: bf16 WMMA + Tensor Data Mover staging.
// ---------------------------------------------------------------------------

typedef __bf16 bf16_t;
typedef __attribute__((ext_vector_type(16))) __bf16 bf16x16;
typedef __attribute__((ext_vector_type(8)))  __bf16 bf16x8;
typedef __attribute__((ext_vector_type(8)))  float  f32x8;
typedef __attribute__((ext_vector_type(4)))  unsigned uint32x4;
typedef __attribute__((ext_vector_type(8)))  int      int32x8;
typedef __attribute__((ext_vector_type(4)))  int      int32x4;

static __device__ __forceinline__ f32x8 wmma_bf16(bf16x16 a, bf16x16 b, f32x8 c) {
  // D = A(16x32) * B(32x16) + C, f32 accumulate
  return __builtin_amdgcn_wmma_f32_16x16x32_bf16(false, a, false, b, (short)0, c,
                                                 false, false);
}

// Load one 16x32 bf16 tile (row-major, leading dim `ld` elements) into the
// wave32 A/B fragment layout:
//   lanes 0-15  : row = lane,    K = 0..7  and 16..23
//   lanes 16-31 : row = lane-16, K = 8..15 and 24..31
static __device__ __forceinline__ bf16x16 frag_ld(const bf16_t* p0, int ld) {
  const int lane = threadIdx.x & 31;
  const int r    = lane & 15;
  const int kb   = (lane >> 4) << 3;    // 0 or 8
  const bf16_t* p = p0 + r * ld + kb;
  bf16x8 a = *(const bf16x8*)(p);
  bf16x8 b = *(const bf16x8*)(p + 16);
  return __builtin_shufflevector(a, b, 0, 1, 2, 3, 4, 5, 6, 7,
                                 8, 9, 10, 11, 12, 13, 14, 15);
}

// ---------------------------------------------------------------------------
// Tensor Data Mover: DMA a 2D bf16 tile (rows x 32 elems, global row stride
// `gstride` elems) into LDS with a 40-element pitch. Padding descriptor:
// pad after every 16 DWORDs (64B row) by 4 DWORDs (16B) -> pitch 80B.
// Descriptor packing per CDNA5 ISA section 8.3/8.4 (D# groups 0 and 1).
// Wave-level op, EXEC ignored; completion tracked on TENSORcnt.
// ---------------------------------------------------------------------------
static __device__ __forceinline__ void tdm_load_tile(const bf16_t* gptr,
                                                     unsigned lds_byte_addr,
                                                     unsigned rows,
                                                     unsigned gstride) {
  unsigned long long ga = (unsigned long long)(size_t)gptr;
  uint32x4 g0;
  g0[0] = 1u;                                          // count=1 (valid user D#)
  g0[1] = lds_byte_addr;                               // lds_addr
  g0[2] = (unsigned)(ga & 0xffffffffu);                // global_addr[31:0]
  g0[3] = (unsigned)((ga >> 32) & 0x01ffffffu)         // global_addr[56:32]
          | (2u << 30);                                // type=2 ("image")
  int32x8 g1;
  // data_size=1 (2B elems), pad_enable, pad_interval=3 (16 DW), pad_amount=3 (4 DW)
  g1[0] = (int)((1u << 16) | (1u << 20) | (3u << 22) | (3u << 25));
  unsigned td0 = gstride;                              // tensor_dim0 (elems)
  unsigned td1 = 1u << 20;                             // tensor_dim1 (huge)
  g1[1] = (int)((td0 & 0xffffu) << 16);                // atomic_addr=0 | td0 lo16
  g1[2] = (int)(((td0 >> 16) & 0xffffu) | ((td1 & 0xffffu) << 16));
  g1[3] = (int)(((td1 >> 16) & 0xffffu) | (32u << 16));// td1 hi16 | tile_dim0=32
  g1[4] = (int)(rows & 0xffffu);                       // tile_dim1=rows, tile_dim2=0
  g1[5] = (int)gstride;                                // tensor_dim0_stride lo32
  g1[6] = 0;                                           // stride0 hi16 | stride1 lo
  g1[7] = 0;
  int32x4 z4 = {0, 0, 0, 0};
#if defined(__clang_major__) && (__clang_major__ >= 23)
  int32x8 z8 = {0, 0, 0, 0, 0, 0, 0, 0};
  __builtin_amdgcn_tensor_load_to_lds(g0, g1, z4, z4, z8, 0);
#else
  __builtin_amdgcn_tensor_load_to_lds(g0, g1, z4, z4, 0);
#endif
}

// ---------------------------------------------------------------------------
// Elementwise f32 -> bf16 (with optional scale)
// ---------------------------------------------------------------------------
__global__ __launch_bounds__(256) void f2b_kernel(const float* __restrict__ in,
                                                  bf16_t* __restrict__ out,
                                                  float scale, long long n) {
  long long i = (long long)blockIdx.x * 256 + threadIdx.x;
  if (i < n) out[i] = (bf16_t)(in[i] * scale);
}

// ---------------------------------------------------------------------------
// GEMM: C[m,n] = sum_k A[m,k] * B[n,k]   (A: MxK bf16, B: NxK bf16, C: f32)
// mode 0: C row-major [M,N]
// mode 1: scatter to (B,H,L,128) layout: m=(b,l) n=(h,d)
// block: 256 threads = 8 waves as 2(M) x 4(N); block tile 64x128, K-step 32.
// A/B tiles staged into LDS by the Tensor Data Mover, double buffered:
// wave 0 issues the next tile's 2 TDM ops, then everyone waits TENSORcnt<=2
// (TDM ops complete in order per wave => current tile is resident).
// ---------------------------------------------------------------------------
__global__ __launch_bounds__(256) void gemm_bf16_nt(
    const bf16_t* __restrict__ A, const bf16_t* __restrict__ Bw,
    float* __restrict__ C, int M, int N, int K, int mode, int Lc) {
  __shared__ __align__(16) bf16_t sA[2][64 * 40];    // pitch 40 via TDM padding
  __shared__ __align__(16) bf16_t sB[2][128 * 40];

  const int tid = threadIdx.x;
  const int m0  = blockIdx.y * 64;
  const int n0  = blockIdx.x * 128;
  const int w   = tid >> 5;
  const int wm  = (w & 1) * 32;
  const int wn  = (w >> 1) * 32;
  const bool dma_wave = (w == 0);

  f32x8 acc00 = {}, acc01 = {}, acc10 = {}, acc11 = {};

  const bf16_t* Abase = A + (size_t)m0 * K;
  const bf16_t* Bbase = Bw + (size_t)n0 * K;

  if (dma_wave) {
    tdm_load_tile(Abase, (unsigned)(size_t)&sA[0][0], 64, (unsigned)K);
    tdm_load_tile(Bbase, (unsigned)(size_t)&sB[0][0], 128, (unsigned)K);
  }

  int buf = 0;
  for (int k0 = 0; k0 < K; k0 += 32) {
    const bool more = (k0 + 32 < K);
    if (dma_wave && more) {
      tdm_load_tile(Abase + k0 + 32, (unsigned)(size_t)&sA[buf ^ 1][0], 64,
                    (unsigned)K);
      tdm_load_tile(Bbase + k0 + 32, (unsigned)(size_t)&sB[buf ^ 1][0], 128,
                    (unsigned)K);
    }
    if (more) {
      __builtin_amdgcn_s_wait_tensorcnt(2);   // current tile done (in-order TDM)
    } else {
      __builtin_amdgcn_s_wait_tensorcnt(0);
    }
    __syncthreads();

    bf16x16 a0 = frag_ld(&sA[buf][wm * 40], 40);
    bf16x16 a1 = frag_ld(&sA[buf][(wm + 16) * 40], 40);
    bf16x16 b0 = frag_ld(&sB[buf][wn * 40], 40);
    bf16x16 b1 = frag_ld(&sB[buf][(wn + 16) * 40], 40);
    acc00 = wmma_bf16(a0, b0, acc00);
    acc01 = wmma_bf16(a0, b1, acc01);
    acc10 = wmma_bf16(a1, b0, acc10);
    acc11 = wmma_bf16(a1, b1, acc11);
    __syncthreads();
    buf ^= 1;
  }

  const int lane = tid & 31;
  const int colb = lane & 15;
  const int roff = (lane >> 4) * 8;
  const int Hn   = N >> 7;

  auto store_acc = [&](const f32x8& acc, int mrow0, int ncol0) {
#pragma unroll
    for (int j = 0; j < 8; ++j) {
      int   row = m0 + mrow0 + roff + j;
      int   col = n0 + ncol0 + colb;
      float val = acc[j];
      if (mode == 0) {
        C[(size_t)row * N + col] = val;
      } else {
        int b = row / Lc, l = row - b * Lc;
        int h = col >> 7, d = col & 127;
        C[(((size_t)b * Hn + h) * (size_t)Lc + l) * 128 + d] = val;
      }
    }
  };
  store_acc(acc00, wm, wn);
  store_acc(acc01, wm, wn + 16);
  store_acc(acc10, wm + 16, wn);
  store_acc(acc11, wm + 16, wn + 16);
}

// ---------------------------------------------------------------------------
// RoPE in place on (BH, L, 128) f32. One thread per (even,odd) pair.
// ---------------------------------------------------------------------------
__global__ __launch_bounds__(256) void rope_kernel(float* __restrict__ x,
                                                   int L, long long npairs) {
  long long i = (long long)blockIdx.x * 256 + threadIdx.x;
  if (i >= npairs) return;
  int       dp   = (int)(i & 63);
  long long rowi = i >> 6;  // (bh*L + l)
  int       l    = (int)(rowi % L);
  // freqs = 10000^(-2*dp/128); ln(1e4)/64 = 0.14391157
  float freq  = __expf(-0.14391157f * (float)dp);
  float theta = (float)l * freq;
  float s, c;
  __sincosf(theta, &s, &c);
  float* p  = x + rowi * 128 + dp * 2;
  float  x1 = p[0], x2 = p[1];
  p[0] = x1 * c - x2 * s;
  p[1] = x1 * s + x2 * c;
}

// ---------------------------------------------------------------------------
// V transpose + bf16 pack: vt[bh][d][l] = (bf16) v[bh][l][d]
// ---------------------------------------------------------------------------
__global__ __launch_bounds__(256) void vtrans_kernel(const float* __restrict__ v,
                                                     bf16_t* __restrict__ vt,
                                                     int L, long long n) {
  long long i = (long long)blockIdx.x * 256 + threadIdx.x;
  if (i >= n) return;
  int       l  = (int)(i % L);
  long long r  = i / L;
  int       d  = (int)(r & 127);
  int       bh = (int)(r >> 7);
  vt[i] = (bf16_t)v[((size_t)bh * L + l) * 128 + d];
}

// ---------------------------------------------------------------------------
// Causal flash attention.
//  qb: (BH,L,128) bf16, pre-scaled by 1/sqrt(128)
//  kb: (BH,L,128) bf16 (roped)
//  vt: (BH,128,L) bf16 (transposed)
//  vh: (B,L,H*128) f32 output
// block = 128 threads = 4 waves; each wave owns 16 query rows; key tiles of 32.
// ---------------------------------------------------------------------------
__global__ __launch_bounds__(128) void attn_kernel(
    const bf16_t* __restrict__ qb, const bf16_t* __restrict__ kb,
    const bf16_t* __restrict__ vt, float* __restrict__ vh, int Hn, int L) {
  __shared__ __align__(16) bf16_t pls[4][16][40];  // per-wave P tile, pitch 40

  const int tid  = threadIdx.x;
  const int w    = tid >> 5;
  const int lane = tid & 31;
  const int colb = lane & 15;
  const int roff = (lane >> 4) * 8;
  const int bh   = blockIdx.y;
  const int b    = bh / Hn;
  const int h    = bh - b * Hn;
  const int q0   = blockIdx.x * 64 + w * 16;
  const int Dm   = Hn << 7;

  // Q fragments for all four 32-wide chunks of Dh=128
  const bf16_t* qbase = qb + ((size_t)bh * L + q0) * 128;
  bf16x16 qf[4];
#pragma unroll
  for (int dsI = 0; dsI < 4; ++dsI) qf[dsI] = frag_ld(qbase + dsI * 32, 128);

  f32x8 o[8];
#pragma unroll
  for (int d = 0; d < 8; ++d) o[d] = (f32x8){};
  float mrow[8], lrow[8];
#pragma unroll
  for (int j = 0; j < 8; ++j) { mrow[j] = -1e30f; lrow[j] = 0.f; }

  const int nkt = (q0 >> 5) + 1;  // causal: key tiles covering keys <= q0+15
  for (int kt = 0; kt < nkt; ++kt) {
    const int     kcol0 = kt * 32;
    const bf16_t* kbase = kb + ((size_t)bh * L + kcol0) * 128;
    f32x8 s0 = {}, s1 = {};
#pragma unroll
    for (int dsI = 0; dsI < 4; ++dsI) {
      bf16x16 k0f = frag_ld(kbase + dsI * 32, 128);             // keys +0..15
      bf16x16 k1f = frag_ld(kbase + 16 * 128 + dsI * 32, 128);  // keys +16..31
      s0 = wmma_bf16(qf[dsI], k0f, s0);
      s1 = wmma_bf16(qf[dsI], k1f, s1);
    }

    // causal mask + online softmax + stage P tile into LDS (A-frag layout)
#pragma unroll
    for (int j = 0; j < 8; ++j) {
      int   qrow = q0 + roff + j;
      int   c0   = kcol0 + colb;
      float v0   = s0[j] + ((c0 > qrow) ? -1e9f : 0.f);
      float v1   = s1[j] + ((c0 + 16 > qrow) ? -1e9f : 0.f);
      float t    = fmaxf(v0, v1);
      t = fmaxf(t, __shfl_xor(t, 1, 16));
      t = fmaxf(t, __shfl_xor(t, 2, 16));
      t = fmaxf(t, __shfl_xor(t, 4, 16));
      t = fmaxf(t, __shfl_xor(t, 8, 16));
      float mnew  = fmaxf(mrow[j], t);
      float alpha = __expf(mrow[j] - mnew);
      mrow[j]     = mnew;
      float p0 = __expf(v0 - mnew);
      float p1 = __expf(v1 - mnew);
      float rs = p0 + p1;
      rs += __shfl_xor(rs, 1, 16);
      rs += __shfl_xor(rs, 2, 16);
      rs += __shfl_xor(rs, 4, 16);
      rs += __shfl_xor(rs, 8, 16);
      lrow[j] = lrow[j] * alpha + rs;
#pragma unroll
      for (int d = 0; d < 8; ++d) o[d][j] *= alpha;
      pls[w][roff + j][colb]      = (bf16_t)p0;
      pls[w][roff + j][colb + 16] = (bf16_t)p1;
    }
    // wave-private LDS round-trip: make the b16 stores visible to our reads
    asm volatile("s_wait_dscnt 0" ::: "memory");

    bf16x16       pf    = frag_ld(&pls[w][0][0], 40);
    const bf16_t* vbase = vt + (size_t)bh * 128 * L + kcol0;
#pragma unroll
    for (int d = 0; d < 8; ++d) {
      bf16x16 vf = frag_ld(vbase + (size_t)(d * 16) * L, L);
      o[d] = wmma_bf16(pf, vf, o[d]);
    }
  }

  // epilogue: normalize and scatter to (B, L, H*128)
#pragma unroll
  for (int j = 0; j < 8; ++j) {
    int    qrow = q0 + roff + j;
    float  inv  = 1.f / lrow[j];
    float* outp = vh + ((size_t)b * L + qrow) * Dm + h * 128;
#pragma unroll
    for (int d = 0; d < 8; ++d) outp[d * 16 + colb] = o[d][j] * inv;
  }
}

// ---------------------------------------------------------------------------
// Host-side orchestration
// ---------------------------------------------------------------------------
extern "C" void kernel_launch(void* const* d_in, const int* in_sizes, int n_in,
                              void* d_out, int out_size, void* d_ws,
                              size_t ws_size, hipStream_t stream) {
  (void)in_sizes; (void)n_in; (void)out_size; (void)ws_size;
  const float* x  = (const float*)d_in[0];
  // d_in[1] is the causal mask; we regenerate causality analytically.
  const float* Wq = (const float*)d_in[2];
  const float* Wk = (const float*)d_in[3];
  const float* Wv = (const float*)d_in[4];
  const float* Wo = (const float*)d_in[5];

  const int       Bn = 2, L = 2048, D = 2048, Hn = 16;
  const long long NE = (long long)Bn * L * D;  // 8,388,608
  const long long NW = (long long)D * D;       // 4,194,304
  const int       M  = Bn * L;                 // 4096

  char*  ws  = (char*)d_ws;
  size_t off = 0;
  auto   wsAlloc = [&](size_t bytes) {
    char* p = ws + off;
    off += (bytes + 255) & ~(size_t)255;
    return p;
  };
  bf16_t* xb  = (bf16_t*)wsAlloc(NE * 2);
  bf16_t* wqb = (bf16_t*)wsAlloc(NW * 2);
  bf16_t* wkb = (bf16_t*)wsAlloc(NW * 2);
  bf16_t* wvb = (bf16_t*)wsAlloc(NW * 2);
  bf16_t* wob = (bf16_t*)wsAlloc(NW * 2);
  float*  qf  = (float*)wsAlloc(NE * 4);
  bf16_t* qbb = (bf16_t*)wsAlloc(NE * 2);
  bf16_t* kbb = (bf16_t*)wsAlloc(NE * 2);
  bf16_t* vtb = (bf16_t*)wsAlloc(NE * 2);
  float*  vhf = (float*)wsAlloc(NE * 4);
  bf16_t* vhb = (bf16_t*)wsAlloc(NE * 2);

  float* out_o = (float*)d_out;
  float* out_k = out_o + NE;
  float* out_v = out_k + NE;

  const int gbE = (int)((NE + 255) / 256);
  const int gbW = (int)((NW + 255) / 256);

  // 1. bf16 conversions of activations and weights
  hipLaunchKernelGGL(f2b_kernel, dim3(gbE), dim3(256), 0, stream, x, xb, 1.f, NE);
  hipLaunchKernelGGL(f2b_kernel, dim3(gbW), dim3(256), 0, stream, Wq, wqb, 1.f, NW);
  hipLaunchKernelGGL(f2b_kernel, dim3(gbW), dim3(256), 0, stream, Wk, wkb, 1.f, NW);
  hipLaunchKernelGGL(f2b_kernel, dim3(gbW), dim3(256), 0, stream, Wv, wvb, 1.f, NW);
  hipLaunchKernelGGL(f2b_kernel, dim3(gbW), dim3(256), 0, stream, Wo, wob, 1.f, NW);

  // 2. QKV projections (scatter straight into (B,H,L,Dh) layout)
  dim3 ggrid(D / 128, M / 64);
  hipLaunchKernelGGL(gemm_bf16_nt, ggrid, dim3(256), 0, stream,
                     xb, wqb, qf, M, D, D, 1, L);
  hipLaunchKernelGGL(gemm_bf16_nt, ggrid, dim3(256), 0, stream,
                     xb, wkb, out_k, M, D, D, 1, L);
  hipLaunchKernelGGL(gemm_bf16_nt, ggrid, dim3(256), 0, stream,
                     xb, wvb, out_v, M, D, D, 1, L);

  // 3. RoPE on q and k (in place; k stays in d_out as a returned tensor)
  const long long npairs = (long long)Bn * Hn * L * 64;
  const int       gbR    = (int)((npairs + 255) / 256);
  hipLaunchKernelGGL(rope_kernel, dim3(gbR), dim3(256), 0, stream, qf, L, npairs);
  hipLaunchKernelGGL(rope_kernel, dim3(gbR), dim3(256), 0, stream, out_k, L, npairs);

  // 4. Pack bf16 operands for attention (softmax scale folded into q)
  hipLaunchKernelGGL(f2b_kernel, dim3(gbE), dim3(256), 0, stream,
                     qf, qbb, 0.08838834764831845f, NE);
  hipLaunchKernelGGL(f2b_kernel, dim3(gbE), dim3(256), 0, stream,
                     out_k, kbb, 1.f, NE);
  hipLaunchKernelGGL(vtrans_kernel, dim3(gbE), dim3(256), 0, stream,
                     out_v, vtb, L, NE);

  // 5. Causal flash attention -> vh (B, L, D)
  hipLaunchKernelGGL(attn_kernel, dim3(L / 64, Bn * Hn), dim3(128), 0, stream,
                     qbb, kbb, vtb, vhf, Hn, L);

  // 6. Output projection
  hipLaunchKernelGGL(f2b_kernel, dim3(gbE), dim3(256), 0, stream,
                     vhf, vhb, 1.f, NE);
  hipLaunchKernelGGL(gemm_bf16_nt, ggrid, dim3(256), 0, stream,
                     vhb, wob, out_o, M, D, D, 0, L);
}
